// EEGGNNWithFeatures_11622181503561
// MI455X (gfx1250) — matrix operand order repaired and checked by reading, not compiled, not run on hardware
//
#include <hip/hip_runtime.h>
#include <hip/hip_bf16.h>
#include <stdint.h>

#define NNODES 50000
#define NEDGES 800000
#define DFEAT  128
#define NGRAPH 2048
#define EPS_BN 1e-5f

typedef __attribute__((ext_vector_type(16))) __bf16       v16bf;
typedef __attribute__((ext_vector_type(8)))  float        v8f;
typedef __attribute__((ext_vector_type(8)))  unsigned int v8u;

__device__ __forceinline__ unsigned short f2bf(float f) {
    unsigned u = __float_as_uint(f);
    u += 0x7FFFu + ((u >> 16) & 1u);          // round-to-nearest-even
    return (unsigned short)(u >> 16);
}

// ---------------------------------------------------------------------------
// Pack fp32 weight W[K=128][N] into wave32 WMMA B-fragment order (bf16).
// Fragment id layout: (((tileN*4 + kb)*32 + lane)*16 + e)
// lane: n = tileN*16 + (lane&15), kh = lane>>4 ; element e -> K = kb*32 + kh*16 + e
// ---------------------------------------------------------------------------
__global__ void k_pack_w(const float* __restrict__ W, unsigned short* __restrict__ F, int N) {
    int id = blockIdx.x * blockDim.x + threadIdx.x;
    int total = (N >> 4) * 2048;              // (N/16 tiles) * 4 kb * 32 lanes * 16 elems
    if (id >= total) return;
    int t   = id >> 11;
    int rem = id & 2047;
    int kb  = rem >> 9;
    rem    &= 511;
    int l   = rem >> 4;
    int e   = rem & 15;
    int kh  = (l >> 4) & 1;
    int n   = (t << 4) + (l & 15);
    int k   = (kb << 5) + (kh << 4) + e;
    F[id] = f2bf(W[k * N + n]);
}

// Fold Linear bias + BatchNorm into scale/shift:  y = dot*scale + shift
__global__ void k_fuse_bn(const float* __restrict__ b, const float* __restrict__ g,
                          const float* __restrict__ be, const float* __restrict__ m,
                          const float* __restrict__ v, float* __restrict__ scale,
                          float* __restrict__ shift, int n) {
    int i = blockIdx.x * blockDim.x + threadIdx.x;
    if (i >= n) return;
    float s = g[i] * rsqrtf(v[i] + EPS_BN);
    scale[i] = s;
    shift[i] = be[i] + (b[i] - m[i]) * s;
}

__global__ void k_copy4(const float4* __restrict__ s, float4* __restrict__ d, int n4) {
    int i = blockIdx.x * blockDim.x + threadIdx.x;
    if (i < n4) d[i] = s[i];
}

__global__ void k_zero(float* __restrict__ p, int n) {
    int i = blockIdx.x * blockDim.x + threadIdx.x;
    if (i < n) p[i] = 0.0f;
}

// agg[dst] += x[src]  -- one thread per (edge, feature); rows are L2-resident.
__global__ void k_scatter(const float* __restrict__ x, const long long* __restrict__ src,
                          const long long* __restrict__ dst, float* __restrict__ agg) {
    int t = blockIdx.x * blockDim.x + threadIdx.x;
    int e = t >> 7;
    if (e >= NEDGES) return;
    int f = t & 127;
    int s = (int)src[e];
    int d = (int)dst[e];
    atomicAdd(&agg[(size_t)d * DFEAT + f], x[(size_t)s * DFEAT + f]);
}

// ---------------------------------------------------------------------------
// Fused  out = ReLU( (in @ W) * scale + shift )
// in: [nrows x 128] fp32 (nrows % 16 == 0), frags: packed bf16 B-fragments,
// out: [nrows x nout].  Block = 256 thr = 8 waves; block covers 16 rows;
// wave w owns column tile w (uniform skip when nout/16 < 8).
// K = 128 -> 4 x v_wmma_f32_16x16x32_bf16, unrolled with immediate-offset
// clause loads for the B fragments.
// ---------------------------------------------------------------------------
__global__ void __launch_bounds__(256) k_gemm_bn_relu(
        const float* __restrict__ in, const unsigned short* __restrict__ frags,
        const float* __restrict__ scale, const float* __restrict__ shift,
        float* __restrict__ out, int nout) {
    __shared__ __align__(16) unsigned short sA[16 * 128];

    int row0 = blockIdx.x << 4;
    // Vectorized fp32 -> bf16 tile fill: 8 elems/thread, one b128 LDS store.
    {
        const float4* src = (const float4*)(in + (size_t)row0 * 128) + threadIdx.x * 2;
        float4 f0 = src[0];
        float4 f1 = src[1];
        uint4 p;
        p.x = (unsigned)f2bf(f0.x) | ((unsigned)f2bf(f0.y) << 16);
        p.y = (unsigned)f2bf(f0.z) | ((unsigned)f2bf(f0.w) << 16);
        p.z = (unsigned)f2bf(f1.x) | ((unsigned)f2bf(f1.y) << 16);
        p.w = (unsigned)f2bf(f1.z) | ((unsigned)f2bf(f1.w) << 16);
        ((uint4*)sA)[threadIdx.x] = p;
    }
    __syncthreads();

    int lane = threadIdx.x & 31;
    int wave = threadIdx.x >> 5;
    int mrow = lane & 15;              // A: row,  B: column (within tile)
    int kh   = (lane >> 4) & 1;        // lane half selects K sub-range

    if (wave < (nout >> 4)) {
        // Base of this wave's B fragments; kb strides by 64 uint4 (1 KB immediates)
        const uint4* pb = (const uint4*)frags + ((size_t)((wave << 7) + lane) << 1);
        v8f acc = {};
#pragma unroll
        for (int kb = 0; kb < 4; ++kb) {
            // A fragment (contiguous, matches packed B): e0..15 -> K=kb*32+kh*16+e
            // (two aligned b128 LDS loads)
            uint4 a0 = *(const uint4*)&sA[mrow * 128 + kb * 32 + kh * 16];
            uint4 a1 = *(const uint4*)&sA[mrow * 128 + kb * 32 + kh * 16 + 8];
            v8u au;
            au[0] = a0.x; au[1] = a0.y; au[2] = a0.z; au[3] = a0.w;
            au[4] = a1.x; au[5] = a1.y; au[6] = a1.z; au[7] = a1.w;
            v16bf A = __builtin_bit_cast(v16bf, au);

            uint4 b0 = pb[kb * 64];
            uint4 b1 = pb[kb * 64 + 1];
            v8u bu;
            bu[0] = b0.x; bu[1] = b0.y; bu[2] = b0.z; bu[3] = b0.w;
            bu[4] = b1.x; bu[5] = b1.y; bu[6] = b1.z; bu[7] = b1.w;
            v16bf B = __builtin_bit_cast(v16bf, bu);

            acc = __builtin_amdgcn_wmma_f32_16x16x32_bf16(
                false, A, false, B, (short)0, acc, false, false);
        }
        // Epilogue per C/D layout: VGPR r -> M = r + 8*(lane>=16), N = lane&15.
        // nrows % 16 == 0 for every launch -> no bounds checks, straight stores.
        int   ncol = (wave << 4) + mrow;
        float s    = scale[ncol];
        float sh   = shift[ncol];
        float* po  = out + (size_t)(row0 + (kh << 3)) * nout + ncol;
#pragma unroll
        for (int r = 0; r < 8; ++r) {
            float y = acc[r] * s + sh;
            po[(size_t)r * nout] = y > 0.0f ? y : 0.0f;
        }
    }
}

// Global mean-pool: accumulate sums + counts with atomics (batch is graph id).
__global__ void k_pool(const float* __restrict__ h, const long long* __restrict__ batch,
                       float* __restrict__ sums, float* __restrict__ cnt) {
    int t = blockIdx.x * blockDim.x + threadIdx.x;
    int node = t >> 7;
    if (node >= NNODES) return;
    int f = t & 127;
    int g = (int)batch[node];
    atomicAdd(&sums[(size_t)g * DFEAT + f], h[(size_t)node * DFEAT + f]);
    if (f == 0) atomicAdd(&cnt[g], 1.0f);
}

__global__ void k_pool_fin(float* __restrict__ sums, const float* __restrict__ cnt) {
    int t = blockIdx.x * blockDim.x + threadIdx.x;
    if (t >= NGRAPH * DFEAT) return;
    int g = t >> 7;
    sums[t] = sums[t] / fmaxf(cnt[g], 1.0f);
}

// logits[g][c] = h[g,:64] . wc2[:,c] + bc2[c]
__global__ void k_logits(const float* __restrict__ h, const float* __restrict__ wc2,
                         const float* __restrict__ bc2, float* __restrict__ out) {
    int t = blockIdx.x * blockDim.x + threadIdx.x;
    if (t >= NGRAPH * 2) return;
    int g = t >> 1, c = t & 1;
    float acc = bc2[c];
#pragma unroll
    for (int j = 0; j < 64; ++j) acc += h[(size_t)g * 64 + j] * wc2[j * 2 + c];
    out[t] = acc;
}

// ---------------------------------------------------------------------------
extern "C" void kernel_launch(void* const* d_in, const int* in_sizes, int n_in,
                              void* d_out, int out_size, void* d_ws, size_t ws_size,
                              hipStream_t stream) {
    const float*     x     = (const float*)d_in[0];
    const long long* eidx  = (const long long*)d_in[1];
    const long long* batch = (const long long*)d_in[2];
    const long long* src   = eidx;
    const long long* dst   = eidx + NEDGES;

    const float* W[5]  = { (const float*)d_in[3],  (const float*)d_in[9],
                           (const float*)d_in[15], (const float*)d_in[21],
                           (const float*)d_in[27] };          // w11 w12 w21 w22 wc1
    const float* B1[5] = { (const float*)d_in[4],  (const float*)d_in[10],
                           (const float*)d_in[16], (const float*)d_in[22],
                           (const float*)d_in[28] };          // biases
    const float* Gm[5] = { (const float*)d_in[5],  (const float*)d_in[11],
                           (const float*)d_in[17], (const float*)d_in[23],
                           (const float*)d_in[29] };          // gamma
    const float* Be[5] = { (const float*)d_in[6],  (const float*)d_in[12],
                           (const float*)d_in[18], (const float*)d_in[24],
                           (const float*)d_in[30] };          // beta
    const float* Mn[5] = { (const float*)d_in[7],  (const float*)d_in[13],
                           (const float*)d_in[19], (const float*)d_in[25],
                           (const float*)d_in[31] };          // running mean
    const float* Vr[5] = { (const float*)d_in[8],  (const float*)d_in[14],
                           (const float*)d_in[20], (const float*)d_in[26],
                           (const float*)d_in[32] };          // running var
    const float* wc2 = (const float*)d_in[33];
    const float* bc2 = (const float*)d_in[34];

    // ---- workspace layout ----
    const size_t NB = (size_t)NNODES * DFEAT * sizeof(float);   // 25.6 MB
    char* base = (char*)d_ws;
    float*          bufA  = (float*)base;
    float*          bufB  = (float*)(base + NB);
    unsigned short* fw[5];
    size_t off = 2 * NB;
    for (int i = 0; i < 4; ++i) { fw[i] = (unsigned short*)(base + off); off += 128 * 128 * 2; }
    fw[4] = (unsigned short*)(base + off); off += 128 * 64 * 2;
    float* ss = (float*)(base + off);                 // 4*(128+128) + (64+64) floats
    off += (4 * 256 + 128) * sizeof(float);
    float* sums = (float*)(base + off); off += (size_t)NGRAPH * DFEAT * sizeof(float);
    float* cnt  = (float*)(base + off); off += (size_t)NGRAPH * sizeof(float);
    float* hcls = (float*)(base + off); off += (size_t)NGRAPH * 64 * sizeof(float);
    (void)ws_size; (void)in_sizes; (void)n_in; (void)out_size;

    // ---- prep: pack weights, fuse BN ----
    for (int i = 0; i < 4; ++i)
        k_pack_w<<<64, 256, 0, stream>>>(W[i], fw[i], 128);
    k_pack_w<<<32, 256, 0, stream>>>(W[4], fw[4], 64);
    for (int i = 0; i < 4; ++i)
        k_fuse_bn<<<1, 128, 0, stream>>>(B1[i], Gm[i], Be[i], Mn[i], Vr[i],
                                         ss + i * 256, ss + i * 256 + 128, 128);
    k_fuse_bn<<<1, 64, 0, stream>>>(B1[4], Gm[4], Be[4], Mn[4], Vr[4],
                                    ss + 1024, ss + 1024 + 64, 64);

    const int n4       = NNODES * DFEAT / 4;
    const int cpBlocks = (n4 + 255) / 256;
    const int scBlocks = (NEDGES * DFEAT) / 256;
    const int gmBlocks = NNODES / 16;               // 3125, exact

    // ---- conv1 ----
    k_copy4<<<cpBlocks, 256, 0, stream>>>((const float4*)x, (float4*)bufA, n4);
    k_scatter<<<scBlocks, 256, 0, stream>>>(x, src, dst, bufA);
    k_gemm_bn_relu<<<gmBlocks, 256, 0, stream>>>(bufA, fw[0], ss + 0,   ss + 128, bufB, 128);
    k_gemm_bn_relu<<<gmBlocks, 256, 0, stream>>>(bufB, fw[1], ss + 256, ss + 384, bufA, 128);

    // ---- conv2 ----
    k_copy4<<<cpBlocks, 256, 0, stream>>>((const float4*)bufA, (float4*)bufB, n4);
    k_scatter<<<scBlocks, 256, 0, stream>>>(bufA, src, dst, bufB);
    k_gemm_bn_relu<<<gmBlocks, 256, 0, stream>>>(bufB, fw[2], ss + 512, ss + 640, bufA, 128);
    k_gemm_bn_relu<<<gmBlocks, 256, 0, stream>>>(bufA, fw[3], ss + 768, ss + 896, bufB, 128);

    // ---- global mean pool ----
    k_zero<<<(NGRAPH * DFEAT + NGRAPH + 255) / 256, 256, 0, stream>>>(sums, NGRAPH * DFEAT + NGRAPH);
    k_pool<<<(NNODES * DFEAT) / 256 + 1, 256, 0, stream>>>(bufB, batch, sums, cnt);
    k_pool_fin<<<(NGRAPH * DFEAT) / 256, 256, 0, stream>>>(sums, cnt);

    // ---- classifier ----
    k_gemm_bn_relu<<<NGRAPH / 16, 256, 0, stream>>>(sums, fw[4], ss + 1024, ss + 1088,
                                                    hcls, 64);
    k_logits<<<(NGRAPH * 2 + 255) / 256, 256, 0, stream>>>(hcls, wc2, bc2, (float*)d_out);
}
